// GNNEncoder_18769007084367
// MI455X (gfx1250) — compile-verified, matrix operand order checked by reading, not run on hardware
//
#include <hip/hip_runtime.h>
#include <hip/hip_bf16.h>

// ---------------------------------------------------------------------------
// Types for CDNA5 WMMA
// ---------------------------------------------------------------------------
typedef __attribute__((ext_vector_type(16))) _Float16 v16h;
typedef __attribute__((ext_vector_type(8)))  float    v8f;

union V16H { v16h v; _Float16 e[16]; };

// ---------------------------------------------------------------------------
// GEMM: H[rows x 64] = X[rows x 64] * W[64 x 64]  (f16 WMMA, f32 accumulate)
//
// Block = 256 threads = 8 wave32s, one 16-row tile per wave.
// W (64x64) is converted to f16 and staged in LDS once per block, already
// swizzled into the WMMA B-fragment layout (ISA 7.12.2):
//   fragment (kt,nt): lane<16 -> N=lane, elems 0..7 = K0..7, 8..15 = K16..23;
//   lane>=16 -> N=lane-16, K8..15 / K24..31  (K relative to kt*32).
// Each wave then reads a whole fragment as one 32-byte LDS access
// (2x ds_load_b128) and each lane reads its A row with 4x global_load_b128.
// ---------------------------------------------------------------------------
__global__ __launch_bounds__(256)
void gemm64_wmma_kernel(const float* __restrict__ X, const float* __restrict__ Wm,
                        float* __restrict__ H, int tiles) {
  __shared__ V16H sW[8 * 32];                // 8 fragments x 32 lanes x 32B = 8KB

  __builtin_prefetch(Wm, 0, 1);              // global_prefetch_b8 for weights

  // ---- cooperative fill: W -> f16 fragments in LDS (4096 halves) ----
  {
    int t = threadIdx.x;
    #pragma unroll
    for (int it = 0; it < 16; ++it) {
      int f    = t + it * 256;               // 0..4095
      int i    = f & 15;                     // element within fragment lane
      int lane = (f >> 4) & 31;
      int nt   = (f >> 9) & 3;
      int kt   = f >> 11;
      int k    = (i & 7) + (lane >> 4) * 8 + (i >> 3) * 16;
      sW[(kt * 4 + nt) * 32 + lane].e[i] =
          (_Float16)Wm[(size_t)(kt * 32 + k) * 64 + nt * 16 + (lane & 15)];
    }
  }
  __syncthreads();

  int tile = blockIdx.x * 8 + (threadIdx.x >> 5);
  if (tile >= tiles) return;                 // wave-uniform: EXEC all-ones below
  int lane = threadIdx.x & 31;
  int hsel = lane >> 4;                      // 0 or 1
  int rc   = lane & 15;                      // A-row / D-col

  v8f acc[4] = {};
  const float* xrow = X + (size_t)(tile * 16 + rc) * 64;

  #pragma unroll
  for (int kt = 0; kt < 2; ++kt) {
    // A fragment: two contiguous 8-float runs -> 4x global_load_b128
    union { float4 q[4]; float f[16]; } ax;
    const float* abase = xrow + kt * 32 + hsel * 8;
    ax.q[0] = *(const float4*)(abase + 0);
    ax.q[1] = *(const float4*)(abase + 4);
    ax.q[2] = *(const float4*)(abase + 16);
    ax.q[3] = *(const float4*)(abase + 20);
    V16H a;
    #pragma unroll
    for (int i = 0; i < 16; ++i) a.e[i] = (_Float16)ax.f[i];

    #pragma unroll
    for (int nt = 0; nt < 4; ++nt) {
      v16h b = sW[(kt * 4 + nt) * 32 + lane].v;   // 2x ds_load_b128
      // (neg_a, A, neg_b, B, c_mod, C, reuse_a, reuse_b)
      acc[nt] = __builtin_amdgcn_wmma_f32_16x16x32_f16(
          false, a.v, false, b, (short)0, acc[nt], false, false);
    }
  }

  #pragma unroll
  for (int nt = 0; nt < 4; ++nt)
    #pragma unroll
    for (int r = 0; r < 8; ++r)
      H[(size_t)(tile * 16 + r + hsel * 8) * 64 + nt * 16 + rc] = acc[nt][r];
}

// Scalar tail for rows not covered by full 16-row tiles (empty when rows%16==0)
__global__ void gemm64_tail_kernel(const float* __restrict__ X, const float* __restrict__ Wm,
                                   float* __restrict__ H, int row0, int rows) {
  int i = blockIdx.x * blockDim.x + threadIdx.x;
  int nrem = (rows - row0) * 64;
  if (i >= nrem) return;
  int row = row0 + i / 64, col = i % 64;
  float s = 0.f;
  #pragma unroll
  for (int k = 0; k < 64; ++k) s += X[(size_t)row * 64 + k] * Wm[(size_t)k * 64 + col];
  H[(size_t)row * 64 + col] = s;
}

// ---------------------------------------------------------------------------
// Per-node attention scores: hs = h . a_src, hd = h . a_dst  (wave per node)
// ---------------------------------------------------------------------------
__global__ void scores_kernel(const float* __restrict__ h,
                              const float* __restrict__ asrc,
                              const float* __restrict__ adst,
                              float* __restrict__ hs, float* __restrict__ hd, int n) {
  int wave = (int)((blockIdx.x * (size_t)blockDim.x + threadIdx.x) >> 5);
  int lane = threadIdx.x & 31;
  if (wave >= n) return;
  const float* row = h + (size_t)wave * 64;
  float s = row[lane] * asrc[lane] + row[lane + 32] * asrc[lane + 32];
  float d = row[lane] * adst[lane] + row[lane + 32] * adst[lane + 32];
  #pragma unroll
  for (int off = 16; off > 0; off >>= 1) {
    s += __shfl_xor(s, off, 32);
    d += __shfl_xor(d, off, 32);
  }
  if (lane == 0) { hs[wave] = s; hd[wave] = d; }
}

// order-preserving float <-> uint encoding for atomic max
__device__ __forceinline__ unsigned fenc(float f) {
  unsigned u = __float_as_uint(f);
  return (u >> 31) ? ~u : (u | 0x80000000u);
}
__device__ __forceinline__ float fdec(unsigned u) {
  return (u >> 31) ? __uint_as_float(u & 0x7FFFFFFFu) : __uint_as_float(~u);
}

__device__ __forceinline__ void edge_sd(const int* ei, int E, int e, int& s, int& d) {
  if (e < E) { s = ei[e]; d = ei[E + e]; } else { s = d = e - E; }  // self-loops
}

// pass 1: logit = leakyrelu(hs[src]+hd[dst], 0.2); segment max into menc[dst]
__global__ void logit_kernel(const float* __restrict__ hs, const float* __restrict__ hd,
                             const int* __restrict__ ei, int E, int n,
                             float* __restrict__ logits, unsigned* __restrict__ menc) {
  int e = blockIdx.x * blockDim.x + threadIdx.x;
  if (e >= E + n) return;
  int s, d; edge_sd(ei, E, e, s, d);
  float l = hs[s] + hd[d];
  l = l > 0.f ? l : 0.2f * l;
  logits[e] = l;
  atomicMax(&menc[d], fenc(l));
}

// pass 2: ex = exp(l - m[dst]); overwrite logits; segment sum into z[dst]
__global__ void expsum_kernel(float* __restrict__ logits, const unsigned* __restrict__ menc,
                              const int* __restrict__ ei, int E, int n,
                              float* __restrict__ z) {
  int e = blockIdx.x * blockDim.x + threadIdx.x;
  if (e >= E + n) return;
  int s, d; edge_sd(ei, E, e, s, d);
  float ex = __expf(logits[e] - fdec(menc[d]));
  logits[e] = ex;
  atomicAdd(&z[d], ex);
}

// pass 3: agg[dst] += h[src] * (ex / z[dst])   (wave per edge, 2 feats/lane)
__global__ void scatter_kernel(const float* __restrict__ h, const float* __restrict__ exs,
                               const float* __restrict__ z, const int* __restrict__ ei,
                               int E, int n, float* __restrict__ agg) {
  size_t idx = blockIdx.x * (size_t)blockDim.x + threadIdx.x;
  int e = (int)(idx >> 5);
  int lane = threadIdx.x & 31;
  if (e >= E + n) return;
  int s, d; edge_sd(ei, E, e, s, d);
  float alpha = exs[e] / z[d];
  const float* row = h + (size_t)s * 64;
  float* orow = agg + (size_t)d * 64;
  atomicAdd(&orow[lane],      row[lane]      * alpha);
  atomicAdd(&orow[lane + 32], row[lane + 32] * alpha);
}

__global__ void bias_add_kernel(float* __restrict__ p, const float* __restrict__ bias, size_t n) {
  size_t i = blockIdx.x * (size_t)blockDim.x + threadIdx.x;
  if (i < n) p[i] += bias[i & 63];
}

// BN batch statistics: stats[0..63] = sum, stats[64..127] = sumsq
__global__ void bn_stats_kernel(const float* __restrict__ h, int n, float* __restrict__ stats) {
  __shared__ float ssum[256], ssq[256];
  int f = threadIdx.x & 63, rg = threadIdx.x >> 6;     // 64 features x 4 row groups
  float sum = 0.f, sq = 0.f;
  for (int r = blockIdx.x * 4 + rg; r < n; r += gridDim.x * 4) {
    float v = h[(size_t)r * 64 + f];
    sum += v; sq += v * v;
  }
  ssum[threadIdx.x] = sum; ssq[threadIdx.x] = sq;
  __syncthreads();
  if (threadIdx.x < 64) {
    int t = threadIdx.x;
    atomicAdd(&stats[t],      ssum[t] + ssum[t + 64] + ssum[t + 128] + ssum[t + 192]);
    atomicAdd(&stats[64 + t], ssq[t]  + ssq[t + 64]  + ssq[t + 128]  + ssq[t + 192]);
  }
}

// BN normalize (biased var) + LeakyReLU(0.01)
__global__ void bn_norm_kernel(const float* __restrict__ h, const float* __restrict__ stats,
                               const float* __restrict__ gamma, const float* __restrict__ beta,
                               int n, float* __restrict__ out) {
  size_t i = blockIdx.x * (size_t)blockDim.x + threadIdx.x;
  if (i >= (size_t)n * 64) return;
  int f = (int)(i & 63);
  float inv = 1.0f / (float)n;
  float mu  = stats[f] * inv;
  float var = stats[64 + f] * inv - mu * mu;
  float v = (h[i] - mu) * rsqrtf(var + 1e-5f) * gamma[f] + beta[f];
  out[i] = v > 0.f ? v : 0.01f * v;
}

__global__ void final_kernel(const float* __restrict__ x, const float* __restrict__ bias,
                             float* __restrict__ out, size_t n) {
  size_t i = blockIdx.x * (size_t)blockDim.x + threadIdx.x;
  if (i < n) out[i] = 0.5f * (x[i] + out[i] + bias[i & 63]);
}

__global__ void fill_zero_kernel(unsigned* __restrict__ p, size_t n) {
  size_t i = blockIdx.x * (size_t)blockDim.x + threadIdx.x;
  if (i < n) p[i] = 0u;
}

// ---------------------------------------------------------------------------
// Host launcher
// ---------------------------------------------------------------------------
static inline int cdiv(size_t a, size_t b) { return (int)((a + b - 1) / b); }

extern "C" void kernel_launch(void* const* d_in, const int* in_sizes, int n_in,
                              void* d_out, int out_size, void* d_ws, size_t ws_size,
                              hipStream_t stream) {
  const float* x     = (const float*)d_in[0];
  const float* W1    = (const float*)d_in[1];
  const float* asrc1 = (const float*)d_in[2];
  const float* adst1 = (const float*)d_in[3];
  const float* bias1 = (const float*)d_in[4];
  const float* gamma = (const float*)d_in[5];
  const float* beta  = (const float*)d_in[6];
  const float* W2    = (const float*)d_in[7];
  const float* asrc2 = (const float*)d_in[8];
  const float* adst2 = (const float*)d_in[9];
  const float* bias2 = (const float*)d_in[10];
  const int*   ei    = (const int*)d_in[11];

  const int n   = in_sizes[0] / 64;
  const int E   = in_sizes[11] / 2;
  const int tot = E + n;
  const size_t nd = (size_t)n * 64;

  // carve workspace
  char* w = (char*)d_ws;
  float*    h      = (float*)w;    w += nd * 4;
  float*    agg    = (float*)w;    w += nd * 4;
  float*    hs     = (float*)w;    w += (size_t)n * 4;
  float*    hd     = (float*)w;    w += (size_t)n * 4;
  float*    z      = (float*)w;    w += (size_t)n * 4;   // z, menc, stats contiguous
  unsigned* menc   = (unsigned*)w; w += (size_t)n * 4;
  float*    stats  = (float*)w;    w += 128 * 4;
  float*    logits = (float*)w;    w += (size_t)tot * 4;
  float*    outp   = (float*)d_out;

  const int tiles = n / 16;
  const int rem   = n - tiles * 16;

  // ---------------- layer 1 ----------------
  fill_zero_kernel<<<cdiv(nd, 256), 256, 0, stream>>>((unsigned*)agg, nd);
  fill_zero_kernel<<<cdiv((size_t)2 * n + 128, 256), 256, 0, stream>>>((unsigned*)z, (size_t)2 * n + 128);

  gemm64_wmma_kernel<<<cdiv(tiles, 8), 256, 0, stream>>>(x, W1, h, tiles);
  if (rem)
    gemm64_tail_kernel<<<cdiv((size_t)rem * 64, 256), 256, 0, stream>>>(x, W1, h, tiles * 16, n);

  scores_kernel<<<cdiv(n, 8), 256, 0, stream>>>(h, asrc1, adst1, hs, hd, n);
  logit_kernel<<<cdiv(tot, 256), 256, 0, stream>>>(hs, hd, ei, E, n, logits, menc);
  expsum_kernel<<<cdiv(tot, 256), 256, 0, stream>>>(logits, menc, ei, E, n, z);
  scatter_kernel<<<cdiv((size_t)tot * 32, 256), 256, 0, stream>>>(h, logits, z, ei, E, n, agg);
  bias_add_kernel<<<cdiv(nd, 256), 256, 0, stream>>>(agg, bias1, nd);

  bn_stats_kernel<<<512, 256, 0, stream>>>(agg, n, stats);
  bn_norm_kernel<<<cdiv(nd, 256), 256, 0, stream>>>(agg, stats, gamma, beta, n, h);

  // ---------------- layer 2 ----------------
  fill_zero_kernel<<<cdiv((size_t)2 * n, 256), 256, 0, stream>>>((unsigned*)z, (size_t)2 * n);
  fill_zero_kernel<<<cdiv(nd, 256), 256, 0, stream>>>((unsigned*)outp, nd);

  gemm64_wmma_kernel<<<cdiv(tiles, 8), 256, 0, stream>>>(h, W2, agg, tiles);
  if (rem)
    gemm64_tail_kernel<<<cdiv((size_t)rem * 64, 256), 256, 0, stream>>>(h, W2, agg, tiles * 16, n);

  scores_kernel<<<cdiv(n, 8), 256, 0, stream>>>(agg, asrc2, adst2, hs, hd, n);
  logit_kernel<<<cdiv(tot, 256), 256, 0, stream>>>(hs, hd, ei, E, n, logits, menc);
  expsum_kernel<<<cdiv(tot, 256), 256, 0, stream>>>(logits, menc, ei, E, n, z);
  scatter_kernel<<<cdiv((size_t)tot * 32, 256), 256, 0, stream>>>(agg, logits, z, ei, E, n, outp);

  final_kernel<<<cdiv(nd, 256), 256, 0, stream>>>(x, bias2, outp, nd);
}
